// MoELayer_10488310137505
// MI455X (gfx1250) — compile-verified, matrix operand order checked by reading, not run on hardware
//
#include <hip/hip_runtime.h>
#include <hip/hip_bf16.h>
#include <math.h>

// ---------------- problem constants ----------------
#define C_DIM   1024
#define H_DIM   4096
#define E_NUM   8
#define N_TOK   8192          // 4 * 2048
#define TOPK    2

// ---------------- CDNA5 WMMA types ----------------
typedef __attribute__((ext_vector_type(16))) __bf16 v16bf;
typedef __attribute__((ext_vector_type(8)))  float  v8f;

// fp32 -> bf16, round-to-nearest-even
__device__ __forceinline__ unsigned short f2bf(float f) {
    unsigned int u = __float_as_uint(f);
    unsigned int r = u + 0x7FFFu + ((u >> 16) & 1u);
    return (unsigned short)(r >> 16);
}

// pack two bf16 (from two fp32) into one dword
__device__ __forceinline__ unsigned int pack2bf(float lo, float hi) {
    return (unsigned int)f2bf(lo) | ((unsigned int)f2bf(hi) << 16);
}

// Branch-free erf-GELU (Abramowitz–Stegun 7.1.26 minimax erf):
// no EXEC-mask divergence, one v_rcp + one v_exp + FMA chain.
__device__ __forceinline__ float gelu_exact(float v) {
    float z = v * 0.70710678118654752440f;
    float a = fabsf(z);
    float t = __builtin_amdgcn_rcpf(fmaf(0.3275911f, a, 1.0f));
    float poly = t * fmaf(t, fmaf(t, fmaf(t, fmaf(t, 1.061405429f, -1.453152027f),
                                          1.421413741f), -0.284496736f), 0.254829592f);
    float er = copysignf(1.0f - poly * __expf(-z * z), z);
    return 0.5f * v * (1.0f + er);
}

// Load one 16-bit WMMA operand fragment (A or B) from a row-major
// [row][32 k] LDS tile, per the CDNA5 ISA 16-bit A-matrix layout:
//   lanes 0-15  : K = 0..7  (VGPR0-3), K = 16..23 (VGPR4-7)
//   lanes 16-31 : K = 8..15 (VGPR0-3), K = 24..31 (VGPR4-7)
// For the B matrix the "row" is a column of B (K contiguous).
__device__ __forceinline__ v16bf load_frag(const unsigned short* rowp, int half) {
    union { uint4 q[2]; v16bf v; } u;
    u.q[0] = *(const uint4*)(rowp + half * 8);        // K 0..7   / 8..15
    u.q[1] = *(const uint4*)(rowp + 16 + half * 8);   // K 16..23 / 24..31
    return u.v;
}

// =====================================================================
// Kernel 0: zero output + per-launch scratch state (graph-replay safe)
// =====================================================================
__global__ void zero_kernel(float* __restrict__ out, long long total,
                            int* __restrict__ counts, float* __restrict__ meanAcc) {
    long long i = (long long)blockIdx.x * blockDim.x + threadIdx.x;
    if (i < total) out[i] = 0.0f;
    if (i < E_NUM) { counts[i] = 0; meanAcc[i] = 0.0f; }
}

// =====================================================================
// Kernel 1: router — logits, softmax, top-2, expert dispatch lists,
//           balance-loss accumulation
// =====================================================================
__global__ __launch_bounds__(256) void router_kernel(
    const float* __restrict__ x, const float* __restrict__ rw,
    const float* __restrict__ rb,
    int* __restrict__ counts, float* __restrict__ meanAcc,
    int* __restrict__ tokIdx, float* __restrict__ tokW) {

    __shared__ float sProb[E_NUM];
    const int tid = threadIdx.x;
    const int n   = blockIdx.x * 256 + tid;
    if (tid < E_NUM) sProb[tid] = 0.0f;
    __syncthreads();

    float logit[E_NUM];
#pragma unroll
    for (int e = 0; e < E_NUM; ++e) logit[e] = rb[e];

    const float* xr = x + (size_t)n * C_DIM;
    for (int k = 0; k < C_DIM; ++k) {
        float xv = xr[k];
        const float* wrow = rw + (size_t)k * E_NUM;
#pragma unroll
        for (int e = 0; e < E_NUM; ++e) logit[e] = fmaf(xv, wrow[e], logit[e]);
    }

    // softmax over 8
    float mx = logit[0];
#pragma unroll
    for (int e = 1; e < E_NUM; ++e) mx = fmaxf(mx, logit[e]);
    float p[E_NUM], sum = 0.0f;
#pragma unroll
    for (int e = 0; e < E_NUM; ++e) { p[e] = __expf(logit[e] - mx); sum += p[e]; }
    float inv = 1.0f / sum;
#pragma unroll
    for (int e = 0; e < E_NUM; ++e) p[e] *= inv;

    // top-2 (ties -> lower index, matching lax.top_k)
    int e1 = 0;
#pragma unroll
    for (int e = 1; e < E_NUM; ++e) if (p[e] > p[e1]) e1 = e;
    int e2 = (e1 == 0) ? 1 : 0;
#pragma unroll
    for (int e = 0; e < E_NUM; ++e) if (e != e1 && p[e] > p[e2]) e2 = e;

    // append into per-expert dispatch lists
    int r1 = atomicAdd(&counts[e1], 1);
    tokIdx[e1 * N_TOK + r1] = n;  tokW[e1 * N_TOK + r1] = p[e1];
    int r2 = atomicAdd(&counts[e2], 1);
    tokIdx[e2 * N_TOK + r2] = n;  tokW[e2 * N_TOK + r2] = p[e2];

    // balance loss accumulation: block-reduce in LDS, one atomic per expert
#pragma unroll
    for (int e = 0; e < E_NUM; ++e) atomicAdd(&sProb[e], p[e]);
    __syncthreads();
    if (tid < E_NUM) atomicAdd(&meanAcc[tid], sProb[tid]);
}

// =====================================================================
// Kernel 2: fused expert FFN.  grid = (N/16, E), 128 threads (4 waves).
// Block handles a 16-token tile of one expert, all 1024 output cols:
//   out += gate * ( gelu(x @ W1 + b1) @ W2 + b2 )
// Hidden is streamed in 128-wide chunks -> computed exactly once.
// =====================================================================
__global__ __launch_bounds__(128) void moe_ffn_kernel(
    const float* __restrict__ x,
    const float* __restrict__ w1, const float* __restrict__ b1,
    const float* __restrict__ w2, const float* __restrict__ b2,
    const int* __restrict__ counts,
    const int* __restrict__ tokIdx, const float* __restrict__ tokW,
    float* __restrict__ out) {

    const int e   = blockIdx.y;
    const int t   = blockIdx.x;
    const int cnt = counts[e];
    if (t * 16 >= cnt) return;              // uniform early-exit

    // LDS: padded to break bank conflicts on ds_load_b128 fragment reads
    __shared__ __align__(16) unsigned short sA[16][1032];   // x tile, bf16
    __shared__ __align__(16) unsigned short sH[16][136];    // hidden chunk, bf16
    __shared__ __align__(16) unsigned short sB[128][40];    // transposed weight slab
    __shared__ int   sTok[16];
    __shared__ float sW[16];

    const int tid  = threadIdx.x;
    const int wave = tid >> 5;
    const int lane = tid & 31;
    const int half = lane >> 4;             // ISA half-wave split
    const int r    = lane & 15;

    if (tid < 16) {
        int row = t * 16 + tid;
        bool ok = row < cnt;
        sTok[tid] = ok ? tokIdx[e * N_TOK + row] : 0;
        sW[tid]   = ok ? tokW[e * N_TOK + row]   : 0.0f;   // zero-gate padding rows
    }
    __syncthreads();

    // gather + convert x tile (coalesced)
    for (int i = tid; i < 16 * C_DIM; i += 128) {
        int rr = i >> 10, cc = i & (C_DIM - 1);
        sA[rr][cc] = f2bf(x[(size_t)sTok[rr] * C_DIM + cc]);
    }
    __syncthreads();

    const float* W1 = w1 + (size_t)e * C_DIM * H_DIM;
    const float* W2 = w2 + (size_t)e * H_DIM * C_DIM;

    v8f accC[16];                            // 16 N-tiles of 16 -> 256 cols/wave
    const v8f vz = {0.f,0.f,0.f,0.f,0.f,0.f,0.f,0.f};
#pragma unroll
    for (int i = 0; i < 16; ++i) accC[i] = vz;

    for (int hIter = 0; hIter < H_DIM / 128; ++hIter) {
        const int hBase = hIter * 128;

        // ---- GEMM1: hidden[16 x 128]; wave w owns cols [w*32, w*32+32)
        v8f hAcc0, hAcc1;
        {
            float bv0 = b1[e * H_DIM + hBase + wave * 32 + r];
            float bv1 = b1[e * H_DIM + hBase + wave * 32 + 16 + r];
#pragma unroll
            for (int j = 0; j < 8; ++j) { hAcc0[j] = bv0; hAcc1[j] = bv1; }
        }
        for (int kk = 0; kk < C_DIM; kk += 32) {
            const float* src = W1 + (size_t)kk * H_DIM + hBase + tid;
            // prefetch next k-slab into cache (global_prefetch_b8)
            if (kk + 32 < C_DIM)
                __builtin_prefetch(src + (size_t)32 * H_DIM, 0, 1);
            __syncthreads();                 // protect previous sB consumers
            // stage W1[kk..kk+31][hBase + 0..127] transposed -> sB[h][k],
            // packed 8 bf16 at a time -> single ds_store_b128
#pragma unroll
            for (int j0 = 0; j0 < 32; j0 += 8) {
                uint4 q;
                q.x = pack2bf(src[(size_t)(j0 + 0) * H_DIM], src[(size_t)(j0 + 1) * H_DIM]);
                q.y = pack2bf(src[(size_t)(j0 + 2) * H_DIM], src[(size_t)(j0 + 3) * H_DIM]);
                q.z = pack2bf(src[(size_t)(j0 + 4) * H_DIM], src[(size_t)(j0 + 5) * H_DIM]);
                q.w = pack2bf(src[(size_t)(j0 + 6) * H_DIM], src[(size_t)(j0 + 7) * H_DIM]);
                *(uint4*)&sB[tid][j0] = q;
            }
            __syncthreads();
            v16bf a  = load_frag(&sA[r][kk], half);
            v16bf bA = load_frag(&sB[wave * 32 + r][0], half);
            v16bf bB = load_frag(&sB[wave * 32 + 16 + r][0], half);
            hAcc0 = __builtin_amdgcn_wmma_f32_16x16x32_bf16(
                        false, a, false, bA, (short)0, hAcc0, false, false);
            hAcc1 = __builtin_amdgcn_wmma_f32_16x16x32_bf16(
                        false, a, false, bB, (short)0, hAcc1, false, false);
        }

        // ---- branch-free GELU, hidden chunk -> LDS as GEMM2 A-matrix (bf16)
#pragma unroll
        for (int j = 0; j < 8; ++j) {
            int m = j + half * 8;            // C/D layout: M = vgpr + 8*half, N = r
            sH[m][wave * 32 + r]      = f2bf(gelu_exact(hAcc0[j]));
            sH[m][wave * 32 + 16 + r] = f2bf(gelu_exact(hAcc1[j]));
        }
        __syncthreads();

        // ---- GEMM2: out[16 x 1024] += hidden_chunk @ W2[hBase..+127][:]
        for (int kc = 0; kc < 4; ++kc) {
            v16bf aH = load_frag(&sH[r][kc * 32], half);
            const int hG = hBase + kc * 32;
            for (int cc = 0; cc < 8; ++cc) {
                const float* src = W2 + (size_t)hG * C_DIM + cc * 128 + tid;
                // prefetch next c-slab / next h-chunk (global_prefetch_b8)
                if (cc < 7)
                    __builtin_prefetch(src + 128, 0, 1);
                else if (kc < 3)
                    __builtin_prefetch(W2 + (size_t)(hG + 32) * C_DIM + tid, 0, 1);
                __syncthreads();
                // stage W2[hG..hG+31][cc*128 .. +127] transposed -> sB[c][h]
#pragma unroll
                for (int j0 = 0; j0 < 32; j0 += 8) {
                    uint4 q;
                    q.x = pack2bf(src[(size_t)(j0 + 0) * C_DIM], src[(size_t)(j0 + 1) * C_DIM]);
                    q.y = pack2bf(src[(size_t)(j0 + 2) * C_DIM], src[(size_t)(j0 + 3) * C_DIM]);
                    q.z = pack2bf(src[(size_t)(j0 + 4) * C_DIM], src[(size_t)(j0 + 5) * C_DIM]);
                    q.w = pack2bf(src[(size_t)(j0 + 6) * C_DIM], src[(size_t)(j0 + 7) * C_DIM]);
                    *(uint4*)&sB[tid][j0] = q;
                }
                __syncthreads();
                v16bf b0  = load_frag(&sB[(wave * 2 + 0) * 16 + r][0], half);
                v16bf b1f = load_frag(&sB[(wave * 2 + 1) * 16 + r][0], half);
                accC[cc * 2 + 0] = __builtin_amdgcn_wmma_f32_16x16x32_bf16(
                        false, aH, false, b0,  (short)0, accC[cc * 2 + 0], false, false);
                accC[cc * 2 + 1] = __builtin_amdgcn_wmma_f32_16x16x32_bf16(
                        false, aH, false, b1f, (short)0, accC[cc * 2 + 1], false, false);
            }
        }
    }

    // ---- gate-weighted scatter-add into output (+ b2 bias)
    float wv[8]; int tk[8];
#pragma unroll
    for (int j = 0; j < 8; ++j) {
        int m = j + half * 8;
        wv[j] = sW[m];
        tk[j] = sTok[m];
    }
#pragma unroll
    for (int cc = 0; cc < 8; ++cc) {
#pragma unroll
        for (int tt = 0; tt < 2; ++tt) {
            v8f a = accC[cc * 2 + tt];
            int c = cc * 128 + (wave * 2 + tt) * 16 + r;
            float b2v = b2[e * C_DIM + c];
#pragma unroll
            for (int j = 0; j < 8; ++j) {
                float val = wv[j] * (a[j] + b2v);
                atomicAdd(&out[(size_t)tk[j] * C_DIM + c], val);
            }
        }
    }
}

// =====================================================================
// Kernel 3: balance loss finalize
// =====================================================================
__global__ void final_kernel(const float* __restrict__ meanAcc, float* __restrict__ loss) {
    if (threadIdx.x == 0) {
        float s = 0.0f;
#pragma unroll
        for (int e = 0; e < E_NUM; ++e) {
            float m = meanAcc[e] * (1.0f / (float)N_TOK);
            s += m * m;
        }
        *loss = s;
    }
}

// =====================================================================
extern "C" void kernel_launch(void* const* d_in, const int* in_sizes, int n_in,
                              void* d_out, int out_size, void* d_ws, size_t ws_size,
                              hipStream_t stream) {
    const float* x  = (const float*)d_in[0];
    const float* rw = (const float*)d_in[1];
    const float* rb = (const float*)d_in[2];
    const float* w1 = (const float*)d_in[3];
    const float* b1 = (const float*)d_in[4];
    const float* w2 = (const float*)d_in[5];
    const float* b2 = (const float*)d_in[6];
    float* out = (float*)d_out;

    // workspace layout (4-byte units): counts[8] | meanAcc[8] | tokIdx[E*N] | tokW[E*N]
    int*   counts  = (int*)d_ws;
    float* meanAcc = (float*)d_ws + 8;
    int*   tokIdx  = (int*)d_ws + 16;
    float* tokW    = (float*)d_ws + 16 + E_NUM * N_TOK;

    const long long outElems = (long long)N_TOK * C_DIM + 1;   // output + loss slot

    zero_kernel<<<(unsigned)((outElems + 255) / 256), 256, 0, stream>>>(
        out, outElems, counts, meanAcc);

    router_kernel<<<N_TOK / 256, 256, 0, stream>>>(
        x, rw, rb, counts, meanAcc, tokIdx, tokW);

    moe_ffn_kernel<<<dim3(N_TOK / 16, E_NUM), 128, 0, stream>>>(
        x, w1, b1, w2, b2, counts, tokIdx, tokW, out);

    final_kernel<<<1, 32, 0, stream>>>(meanAcc, out + (size_t)N_TOK * C_DIM);
}
